// ElementWiseAttention_49632642072629
// MI455X (gfx1250) — matrix operand order repaired and checked by reading, not compiled, not run on hardware
//
#include <hip/hip_runtime.h>
#include <hip/hip_bf16.h>

#define CDIM 256
#define HEADS 8
#define HD 32
#define NPTS 512
#define NE 128
#define NFULL 65536
#define NRED 50000

typedef __attribute__((ext_vector_type(16))) __bf16 v16bf;
typedef __attribute__((ext_vector_type(8)))  __bf16 v8bf;
typedef __attribute__((ext_vector_type(8)))  float  v8f;
typedef __attribute__((ext_vector_type(4)))  unsigned v4u;
typedef __attribute__((ext_vector_type(8)))  int      v8i;
typedef __attribute__((ext_vector_type(4)))  int      v4i;

// ---------- bf16 helpers (round-to-nearest-even) ----------
__device__ __forceinline__ __bf16 f2bf(float x) {
    unsigned u = __builtin_bit_cast(unsigned, x);
    u += 0x7FFFu + ((u >> 16) & 1u);
    unsigned short h = (unsigned short)(u >> 16);
    return __builtin_bit_cast(__bf16, h);
}
__device__ __forceinline__ float bf2f(__bf16 b) {
    unsigned short h = __builtin_bit_cast(unsigned short, b);
    return __builtin_bit_cast(float, ((unsigned)h) << 16);
}

// ---------- WMMA fragment loaders (wave32 layouts, CDNA5 ISA) ----------
__device__ __forceinline__ v16bf load_frag2(const __bf16* p0, const __bf16* p1) {
    v8bf lo = *(const v8bf*)p0;
    v8bf hi = *(const v8bf*)p1;
    return __builtin_shufflevector(lo, hi, 0, 1, 2, 3, 4, 5, 6, 7,
                                   8, 9, 10, 11, 12, 13, 14, 15);
}
// A 16x32 bf16: lane holds row M=lane&15; K = kb..kb+7 and 16+kb..16+kb+7, kb=(lane>>4)*8
__device__ __forceinline__ v16bf load_afrag(const __bf16* rowp, int lane) {
    int kb = (lane >> 4) << 3;
    return load_frag2(rowp + kb, rowp + 16 + kb);
}
// B 32x16 bf16: lane holds col N=lane&15; K = kb..kb+7 and 16+kb..16+kb+7, kb=(lane>>4)*8
__device__ __forceinline__ v16bf load_bfrag(const __bf16* rowp, int lane) {
    int o = (lane >> 4) << 3;
    return load_frag2(rowp + o, rowp + 16 + o);
}

// ---------- VALU lane crossbar: xor-butterfly max within 16-lane groups ----------
__device__ __forceinline__ float pl16(float v, unsigned s0, unsigned s1) {
    unsigned i = __builtin_bit_cast(unsigned, v);
    i = __builtin_amdgcn_permlane16(i, i, s0, s1, false, false);
    return __builtin_bit_cast(float, i);
}
__device__ __forceinline__ float rowmax16(float x) {
    x = fmaxf(x, pl16(x, 0xFEDCBA98u, 0x76543210u));  // ^8
    x = fmaxf(x, pl16(x, 0x32107654u, 0xBA98FEDCu));  // ^4
    x = fmaxf(x, pl16(x, 0x54761032u, 0xDCFE98BAu));  // ^2
    x = fmaxf(x, pl16(x, 0x67452301u, 0xEFCDAB89u));  // ^1
    return x;
}

// ---------- TDM: stage 64x32 bf16 W tile into LDS (D# per CDNA5 ISA) ----------
__device__ __forceinline__ void tdm_load_wtile(const __bf16* gsrc, unsigned lds_off,
                                               int N, int K) {
    unsigned long long ga = (unsigned long long)(size_t)gsrc;
    v4u g0;
    g0.x = 1u;                                               // count=1, user mode
    g0.y = lds_off;                                          // lds_addr
    g0.z = (unsigned)(ga & 0xFFFFFFFFull);                   // global_addr[31:0]
    g0.w = (unsigned)((ga >> 32) & 0x01FFFFFFull) | (2u << 30); // addr[56:32] | type=2
    v8i g1;
    g1[0] = 0x00010000;                                      // data_size=1 (2 bytes)
    g1[1] = (int)((unsigned)(K & 0xFFFF) << 16);             // tensor_dim0[15:0]
    g1[2] = (int)(((unsigned)(K >> 16) & 0xFFFFu) | ((unsigned)(N & 0xFFFF) << 16));
    g1[3] = (int)(((unsigned)(N >> 16) & 0xFFFFu) | (32u << 16)); // tile_dim0=32
    g1[4] = 64;                                              // tile_dim1=64, tile_dim2=0
    g1[5] = K;                                               // tensor_dim0_stride[31:0]
    g1[6] = 0;
    g1[7] = 0;
    v4i gz = {0, 0, 0, 0};
    v8i gz8 = {0, 0, 0, 0, 0, 0, 0, 0};
    __builtin_amdgcn_tensor_load_to_lds(g0, g1, gz, gz, gz8, 0);
}

// ---------- block reduction over 256 threads ----------
__device__ __forceinline__ float block_sum256(float v) {
    __shared__ float red[8];
#pragma unroll
    for (int o = 16; o > 0; o >>= 1) v += __shfl_xor(v, o, 32);
    if ((threadIdx.x & 31) == 0) red[threadIdx.x >> 5] = v;
    __syncthreads();
    float s = red[0] + red[1] + red[2] + red[3] + red[4] + red[5] + red[6] + red[7];
    __syncthreads();
    return s;
}

// ---------- 1: float -> bf16 convert ----------
__global__ void convert_bf16_kernel(const float* __restrict__ in, __bf16* __restrict__ out, int n) {
    int i = blockIdx.x * 256 + threadIdx.x;
    if (i < n) out[i] = f2bf(in[i]);
}

// ---------- 2: gather + layernorm1 + pos normalize ----------
__global__ __launch_bounds__(256)
void gather_ln1_kernel(const float* __restrict__ x, const float* __restrict__ pos,
                       const int* __restrict__ r2f,
                       const float* __restrict__ w, const float* __restrict__ b,
                       __bf16* __restrict__ xn, float* __restrict__ pf) {
    int f = blockIdx.x, c = threadIdx.x;
    int rr = r2f[f];
    float v = x[(size_t)rr * CDIM + c];
    float mean = block_sum256(v) * (1.0f / CDIM);
    float dv = v - mean;
    float var = block_sum256(dv * dv) * (1.0f / CDIM);
    float y = dv * rsqrtf(var + 1e-5f) * w[c] + b[c];
    xn[(size_t)f * CDIM + c] = f2bf(y);
    if (c < 3) {
        const float mn[3] = {-10.f, -1.f, 0.f};
        const float mx[3] = {25.f, 5.f, 2.f};
        pf[(size_t)f * 3 + c] = (pos[(size_t)rr * 3 + c] - mn[c]) / (mx[c] - mn[c]);
    }
}

// ---------- 3: WMMA GEMM  C[M,N] = A[M,K] * W[N,K]^T, W tile staged by TDM ----------
template <bool STORE_BF16>
__global__ __launch_bounds__(256)
void gemm_bf16_wmma(const __bf16* __restrict__ A, const __bf16* __restrict__ W,
                    float* __restrict__ Cf, __bf16* __restrict__ Cb,
                    int M, int N, int K) {
    __shared__ __bf16 wtile[2][64 * 32];          // double-buffered 4KB W tiles
    int lane = threadIdx.x & 31, wave = threadIdx.x >> 5;
    int m0 = blockIdx.x * 128 + wave * 16;
    int n0 = blockIdx.y * 64;
    int mrow = m0 + (lane & 15);
    if (mrow >= M) mrow = M - 1;                  // clamp loads; stores guarded
    const __bf16* Ap = A + (size_t)mrow * K;
    const int ksteps = K >> 5;

    if (wave == 0)
        tdm_load_wtile(W + (size_t)n0 * K, (unsigned)(size_t)&wtile[0][0], N, K);

    v8f acc[4] = {};
    for (int s = 0; s < ksteps; ++s) {
        int k0 = s << 5;
        if (s + 1 < ksteps) {
            if (wave == 0)
                tdm_load_wtile(W + (size_t)n0 * K + (k0 + 32),
                               (unsigned)(size_t)&wtile[(s + 1) & 1][0], N, K);
            __builtin_amdgcn_s_wait_tensorcnt(1);  // oldest (current) tile done
        } else {
            __builtin_amdgcn_s_wait_tensorcnt(0);
        }
        __syncthreads();
        __builtin_prefetch(Ap + k0 + 64, 0, 1);    // global_prefetch_b8 for next A block
        v16bf a = load_afrag(Ap + k0, lane);
        const __bf16* wt = &wtile[s & 1][0];
#pragma unroll
        for (int t = 0; t < 4; ++t) {
            v16bf bb = load_bfrag(wt + (size_t)(t * 16 + (lane & 15)) * 32, lane);
            acc[t] = __builtin_amdgcn_wmma_f32_16x16x32_bf16(
                false, a, false, bb, (short)0, acc[t], false, false);
        }
        __syncthreads();                           // all reads done before next TDM overwrite
    }
    int rbase = m0 + ((lane >> 4) << 3);
#pragma unroll
    for (int t = 0; t < 4; ++t) {
        int col = n0 + t * 16 + (lane & 15);
#pragma unroll
        for (int r = 0; r < 8; ++r) {
            int row = rbase + r;
            if (row < M) {
                size_t o = (size_t)row * N + col;
                if (STORE_BF16) Cb[o] = f2bf(acc[t][r]); else Cf[o] = acc[t][r];
            }
        }
    }
}

// ---------- 4: RoPE in place on q,k (bf16); zero dims 30..31; fold 1/sqrt(30) into q ----------
__global__ __launch_bounds__(256)
void rope_kernel(__bf16* __restrict__ q, __bf16* __restrict__ k,
                 const float* __restrict__ pf, float qscale) {
    int f = blockIdx.x, c = threadIdx.x;
    int d = c & 31;
    size_t idx = (size_t)f * CDIM + c;
    float qo = 0.f, ko = 0.f;
    if (d < 30) {
        int chunk = d / 10, jj = d % 10;
        bool first = jj < 5;
        int j = first ? jj : jj - 5;
        float ts = __powf(100.f, (float)j * 0.2f);
        float th = pf[(size_t)f * 3 + chunk] / ts;
        float sn = __sinf(th), cs = __cosf(th);
        size_t pidx = (size_t)f * CDIM + (first ? c + 5 : c - 5);
        float qv = bf2f(q[idx]), kv = bf2f(k[idx]);
        float qp = bf2f(q[pidx]), kp = bf2f(k[pidx]);
        if (first) { qo = qv * cs - qp * sn; ko = kv * cs - kp * sn; }
        else       { qo = qv * cs + qp * sn; ko = kv * cs + kp * sn; }
    }
    __syncthreads();   // all reads before any in-place write
    q[idx] = f2bf(qo * qscale);
    k[idx] = f2bf(ko);
}

// ---------- 5: V transpose: vt[(e*H+h)*32+d][n] = v[e*512+n][h*32+d] ----------
__global__ __launch_bounds__(256)
void vtrans_kernel(const __bf16* __restrict__ v, __bf16* __restrict__ vt) {
    int f = blockIdx.x, c = threadIdx.x;
    int e = f >> 9, n = f & 511, h = c >> 5, d = c & 31;
    vt[((size_t)(e * HEADS + h) * 32 + d) * NPTS + n] = v[(size_t)f * CDIM + c];
}

// ---------- 6: flash attention, per-(e,h), one 16-row Q tile per wave ----------
__global__ __launch_bounds__(256)
void attn_fa_kernel(const __bf16* __restrict__ q, const __bf16* __restrict__ k,
                    const __bf16* __restrict__ vt, __bf16* __restrict__ out) {
    __shared__ __bf16 pbuf[8][16 * 32];           // private P tile per wave
    int lane = threadIdx.x & 31, wave = threadIdx.x >> 5;
    int blk = blockIdx.x;
    int qblk = blk & 3;                            // 4 q-blocks of 128 rows
    int h = (blk >> 2) & 7;
    int e = blk >> 5;
    int q0 = qblk * 128 + wave * 16;
    size_t rowbase = (size_t)e * NPTS;
    const __bf16* qp = q + (rowbase + q0 + (lane & 15)) * CDIM + h * HD;
    v16bf qa = load_afrag(qp, lane);               // Q fragment loaded once (K=32)
    v16bf onesb;
    {
        __bf16 one = f2bf(1.0f);
#pragma unroll
        for (int i = 0; i < 16; ++i) onesb[i] = one;
    }
    float mrun[8], srun[8], corr[8];
#pragma unroll
    for (int r = 0; r < 8; ++r) { mrun[r] = -3.0e38f; srun[r] = 0.f; }
    v8f acc0 = {}, acc1 = {};
    __bf16* pb = &pbuf[wave][0];
    const __bf16* vbase = vt + (size_t)(e * HEADS + h) * 32 * NPTS;

    for (int kv0 = 0; kv0 < NPTS; kv0 += 32) {
        // S tile (16 q x 32 kv): two WMMAs, B = K rows (contiguous head-dim)
        v8f s0 = {}, s1 = {};
        {
            const __bf16* kp0 = k + (rowbase + kv0 + (lane & 15)) * CDIM + h * HD;
            v16bf b0 = load_bfrag(kp0, lane);
            s0 = __builtin_amdgcn_wmma_f32_16x16x32_bf16(false, qa, false, b0, (short)0, s0, false, false);
            const __bf16* kp1 = k + (rowbase + kv0 + 16 + (lane & 15)) * CDIM + h * HD;
            v16bf b1 = load_bfrag(kp1, lane);
            s1 = __builtin_amdgcn_wmma_f32_16x16x32_bf16(false, qa, false, b1, (short)0, s1, false, false);
        }
        // online softmax: row max via v_permlane16 butterflies (no LDS traffic)
#pragma unroll
        for (int r = 0; r < 8; ++r) {
            float xm = rowmax16(fmaxf(s0[r], s1[r]));
            float mnew = fmaxf(mrun[r], xm);
            corr[r] = __expf(mrun[r] - mnew);
            float p0 = __expf(s0[r] - mnew);
            float p1 = __expf(s1[r] - mnew);
            mrun[r] = mnew;
            acc0[r] *= corr[r];
            acc1[r] *= corr[r];
            int prow = r + ((lane >> 4) << 3);
            pb[prow * 32 + (lane & 15)]      = f2bf(p0);
            pb[prow * 32 + 16 + (lane & 15)] = f2bf(p1);
        }
        asm volatile("s_wait_dscnt 0x0" ::: "memory");   // LDS RAW across lanes, same wave
        v16bf pa = load_afrag(pb + (lane & 15) * 32, lane);
        // row sums via WMMA against all-ones B: lands replicated in C layout
        v8f rs = {};
        rs = __builtin_amdgcn_wmma_f32_16x16x32_bf16(false, pa, false, onesb, (short)0, rs, false, false);
#pragma unroll
        for (int r = 0; r < 8; ++r) srun[r] = srun[r] * corr[r] + rs[r];
        // P*V: B = vt rows (contiguous kv)
        const __bf16* vp0 = vbase + (size_t)(lane & 15) * NPTS + kv0;
        v16bf vb0 = load_bfrag(vp0, lane);
        acc0 = __builtin_amdgcn_wmma_f32_16x16x32_bf16(false, pa, false, vb0, (short)0, acc0, false, false);
        const __bf16* vp1 = vbase + (size_t)(16 + (lane & 15)) * NPTS + kv0;
        v16bf vb1 = load_bfrag(vp1, lane);
        acc1 = __builtin_amdgcn_wmma_f32_16x16x32_bf16(false, pa, false, vb1, (short)0, acc1, false, false);
    }
#pragma unroll
    for (int r = 0; r < 8; ++r) {
        float inv = 1.0f / srun[r];
        size_t row = rowbase + q0 + r + ((lane >> 4) << 3);
        out[row * CDIM + h * HD + (lane & 15)]      = f2bf(acc0[r] * inv);
        out[row * CDIM + h * HD + 16 + (lane & 15)] = f2bf(acc1[r] * inv);
    }
}

// ---------- 7: zero ----------
__global__ void zero_kernel(float* __restrict__ p, size_t n) {
    size_t i = (size_t)blockIdx.x * blockDim.x + threadIdx.x;
    size_t stride = (size_t)gridDim.x * blockDim.x;
    for (; i < n; i += stride) p[i] = 0.f;
}

// ---------- 8: segment sums + counts ----------
__global__ __launch_bounds__(256)
void segsum_kernel(const float* __restrict__ o_out, const int* __restrict__ index,
                   const int* __restrict__ r2f,
                   float* __restrict__ sums, float* __restrict__ cnt) {
    int f = blockIdx.x, c = threadIdx.x;
    int seg = index[r2f[f]];
    atomicAdd(&sums[(size_t)seg * CDIM + c], o_out[(size_t)f * CDIM + c]);
    if (c == 0) atomicAdd(&cnt[seg], 1.0f);
}

// ---------- 9: x2 = x + mean[index[r2f[f2r[r]]]] ----------
__global__ __launch_bounds__(256)
void resid_kernel(const float* __restrict__ x, const int* __restrict__ index,
                  const int* __restrict__ r2f, const int* __restrict__ f2r,
                  const float* __restrict__ sums, const float* __restrict__ cnt,
                  float* __restrict__ x2) {
    int r = blockIdx.x, c = threadIdx.x;
    int seg = index[r2f[f2r[r]]];
    float m = sums[(size_t)seg * CDIM + c] / fmaxf(cnt[seg], 1.0f);
    x2[(size_t)r * CDIM + c] = x[(size_t)r * CDIM + c] + m;
}

// ---------- 10: layernorm2 ----------
__global__ __launch_bounds__(256)
void ln2_kernel(const float* __restrict__ x2, const float* __restrict__ w,
                const float* __restrict__ b, __bf16* __restrict__ y) {
    int r = blockIdx.x, c = threadIdx.x;
    float v = x2[(size_t)r * CDIM + c];
    float mean = block_sum256(v) * (1.0f / CDIM);
    float dv = v - mean;
    float var = block_sum256(dv * dv) * (1.0f / CDIM);
    y[(size_t)r * CDIM + c] = f2bf(dv * rsqrtf(var + 1e-5f) * w[c] + b[c]);
}

// ---------- 11: SwiGLU activation ----------
__global__ __launch_bounds__(256)
void act_kernel(const float* __restrict__ h, __bf16* __restrict__ act) {
    int r = blockIdx.x, c = threadIdx.x;
    float a = h[(size_t)r * 2 * CDIM + c];
    float g = h[(size_t)r * 2 * CDIM + CDIM + c];
    float sg = g / (1.f + __expf(-g));
    act[(size_t)r * CDIM + c] = f2bf(a * sg);
}

// ---------- 12: final residual ----------
__global__ __launch_bounds__(256)
void final_add_kernel(const float* __restrict__ x2, const float* __restrict__ t,
                      float* __restrict__ out) {
    int r = blockIdx.x, c = threadIdx.x;
    size_t i = (size_t)r * CDIM + c;
    out[i] = x2[i] + t[i];
}

extern "C" void kernel_launch(void* const* d_in, const int* in_sizes, int n_in,
                              void* d_out, int out_size, void* d_ws, size_t ws_size,
                              hipStream_t stream) {
    const float* x     = (const float*)d_in[0];
    const float* pos   = (const float*)d_in[1];
    const int*   index = (const int*)d_in[2];
    const int*   r2f   = (const int*)d_in[3];
    const int*   f2r   = (const int*)d_in[4];
    const float* n1w   = (const float*)d_in[5];
    const float* n1b   = (const float*)d_in[6];
    const float* qw    = (const float*)d_in[7];
    const float* kw    = (const float*)d_in[8];
    const float* vw    = (const float*)d_in[9];
    const float* ow    = (const float*)d_in[10];
    const float* n2w   = (const float*)d_in[11];
    const float* n2b   = (const float*)d_in[12];
    const float* w1    = (const float*)d_in[13];
    const float* w2    = (const float*)d_in[14];

    char* ws = (char*)d_ws;
    size_t off = 0;
    auto alloc = [&](size_t bytes) { size_t o = off; off += (bytes + 255) & ~(size_t)255; return o; };

    size_t o_wq  = alloc((size_t)CDIM * CDIM * 2);
    size_t o_wk  = alloc((size_t)CDIM * CDIM * 2);
    size_t o_wv  = alloc((size_t)CDIM * CDIM * 2);
    size_t o_wo  = alloc((size_t)CDIM * CDIM * 2);
    size_t o_w1  = alloc((size_t)2 * CDIM * CDIM * 2);
    size_t o_w2  = alloc((size_t)CDIM * CDIM * 2);
    size_t o_xn  = alloc((size_t)NFULL * CDIM * 2);
    size_t o_pf  = alloc((size_t)NFULL * 3 * 4);
    size_t o_qb  = alloc((size_t)NFULL * CDIM * 2);   // qb+kb contiguous: reused as O-proj f32 out
    size_t o_kb  = alloc((size_t)NFULL * CDIM * 2);
    size_t o_vb  = alloc((size_t)NFULL * CDIM * 2);
    size_t o_vt  = alloc((size_t)NFULL * CDIM * 2);
    size_t o_at  = alloc((size_t)NFULL * CDIM * 2);
    size_t o_sum = alloc((size_t)NRED * CDIM * 4);
    size_t o_cnt = alloc((size_t)NRED * 4);
    size_t o_x2  = alloc((size_t)NRED * CDIM * 4);
    size_t o_y   = alloc((size_t)NRED * CDIM * 2);
    size_t o_h   = alloc((size_t)NRED * 2 * CDIM * 4); // ffn2 out (tmp) aliases h after act
    size_t o_act = alloc((size_t)NRED * CDIM * 2);

    __bf16* wqb = (__bf16*)(ws + o_wq);
    __bf16* wkb = (__bf16*)(ws + o_wk);
    __bf16* wvb = (__bf16*)(ws + o_wv);
    __bf16* wob = (__bf16*)(ws + o_wo);
    __bf16* w1b = (__bf16*)(ws + o_w1);
    __bf16* w2b = (__bf16*)(ws + o_w2);
    __bf16* xn  = (__bf16*)(ws + o_xn);
    float*  pf  = (float*)(ws + o_pf);
    __bf16* qb  = (__bf16*)(ws + o_qb);
    __bf16* kb  = (__bf16*)(ws + o_kb);
    __bf16* vb  = (__bf16*)(ws + o_vb);
    __bf16* vt  = (__bf16*)(ws + o_vt);
    __bf16* at  = (__bf16*)(ws + o_at);
    float*  oo  = (float*)(ws + o_qb);    // alias: q/k dead after attention
    float*  sums= (float*)(ws + o_sum);
    float*  cnt = (float*)(ws + o_cnt);
    float*  x2  = (float*)(ws + o_x2);
    __bf16* yb  = (__bf16*)(ws + o_y);
    float*  hb  = (float*)(ws + o_h);
    float*  tmp = (float*)(ws + o_h);     // alias: h dead after act_kernel
    __bf16* actb= (__bf16*)(ws + o_act);

    // 1) weights -> bf16
    convert_bf16_kernel<<<(CDIM * CDIM + 255) / 256, 256, 0, stream>>>(qw, wqb, CDIM * CDIM);
    convert_bf16_kernel<<<(CDIM * CDIM + 255) / 256, 256, 0, stream>>>(kw, wkb, CDIM * CDIM);
    convert_bf16_kernel<<<(CDIM * CDIM + 255) / 256, 256, 0, stream>>>(vw, wvb, CDIM * CDIM);
    convert_bf16_kernel<<<(CDIM * CDIM + 255) / 256, 256, 0, stream>>>(ow, wob, CDIM * CDIM);
    convert_bf16_kernel<<<(2 * CDIM * CDIM + 255) / 256, 256, 0, stream>>>(w1, w1b, 2 * CDIM * CDIM);
    convert_bf16_kernel<<<(CDIM * CDIM + 255) / 256, 256, 0, stream>>>(w2, w2b, CDIM * CDIM);

    // 2) gather + LN1
    gather_ln1_kernel<<<NFULL, 256, 0, stream>>>(x, pos, r2f, n1w, n1b, xn, pf);

    // 3) QKV projections (WMMA + TDM-staged W tiles)
    dim3 gqkv(NFULL / 128, CDIM / 64);
    gemm_bf16_wmma<true><<<gqkv, 256, 0, stream>>>(xn, wqb, nullptr, qb, NFULL, CDIM, CDIM);
    gemm_bf16_wmma<true><<<gqkv, 256, 0, stream>>>(xn, wkb, nullptr, kb, NFULL, CDIM, CDIM);
    gemm_bf16_wmma<true><<<gqkv, 256, 0, stream>>>(xn, wvb, nullptr, vb, NFULL, CDIM, CDIM);

    // 4) RoPE (scale = 1/sqrt(30) folded into q)
    rope_kernel<<<NFULL, 256, 0, stream>>>(qb, kb, pf, 0.18257418583505536f);

    // 5) V transpose for PV fragments
    vtrans_kernel<<<NFULL, 256, 0, stream>>>(vb, vt);

    // 6) flash attention (WMMA)
    attn_fa_kernel<<<NE * HEADS * 4, 256, 0, stream>>>(qb, kb, vt, at);

    // 7) O projection (WMMA) -> f32 (aliases q/k region)
    gemm_bf16_wmma<false><<<gqkv, 256, 0, stream>>>(at, wob, oo, nullptr, NFULL, CDIM, CDIM);

    // 8) segment mean
    zero_kernel<<<2048, 256, 0, stream>>>(sums, (size_t)NRED * CDIM + NRED); // cnt follows sums
    segsum_kernel<<<NFULL, 256, 0, stream>>>(oo, index, r2f, sums, cnt);
    resid_kernel<<<NRED, 256, 0, stream>>>(x, index, r2f, f2r, sums, cnt, x2);

    // 9) LN2 + FFN (WMMA) + final residual
    ln2_kernel<<<NRED, 256, 0, stream>>>(x2, n2w, n2b, yb);
    dim3 gf1((NRED + 127) / 128, (2 * CDIM) / 64);
    gemm_bf16_wmma<false><<<gf1, 256, 0, stream>>>(yb, w1b, hb, nullptr, NRED, 2 * CDIM, CDIM);
    act_kernel<<<NRED, 256, 0, stream>>>(hb, actb);
    dim3 gf2((NRED + 127) / 128, CDIM / 64);
    gemm_bf16_wmma<false><<<gf2, 256, 0, stream>>>(actb, w2b, tmp, nullptr, NRED, CDIM, CDIM);
    final_add_kernel<<<NRED, 256, 0, stream>>>(x2, tmp, (float*)d_out);
}
